// GINE_9088150798515
// MI455X (gfx1250) — matrix operand order repaired and checked by reading, not compiled, not run on hardware
//
#include <hip/hip_runtime.h>

#define NNODES 50000
#define NEDGES 600000
#define NGRAPHS 128
#define DIN 128
#define DEDGE 64
#define NLAYERS 4
#define BN_EPS 1e-5f
#define NF (NNODES * DIN)

typedef __attribute__((ext_vector_type(16))) __bf16 v16bf;
typedef __attribute__((ext_vector_type(8)))  __bf16 v8bf;
typedef __attribute__((ext_vector_type(8)))  float  v8f;
typedef __attribute__((ext_vector_type(4)))  float  v4f;

// ---------------------------------------------------------------------------
// D = A(16x32 bf16) * B(32x16 bf16) + C(16x16 f32), wave32 WMMA
// ---------------------------------------------------------------------------
__device__ __forceinline__ v8f wmma_bf16(v16bf a, v16bf b, v8f c) {
  return __builtin_amdgcn_wmma_f32_16x16x32_bf16(
      /*neg_a=*/false, a, /*neg_b=*/false, b,
      /*c_mod=*/(short)0, c, /*reuse_a=*/false, /*reuse_b=*/false);
}

__device__ __forceinline__ void atomic_add_f32(float* p, float v) {
  unsafeAtomicAdd(p, v);  // global_atomic_add_f32 (L2 atomic unit)
}

// A-operand: lane holds M=lane%16,
// e=0..15 -> K=kb+e; kb includes (lane<16?0:16)+chunk.
__device__ __forceinline__ v16bf pack_a(v4f f0, v4f f1, v4f f2, v4f f3) {
  v16bf a;
  a[0]  = (__bf16)f0[0]; a[1]  = (__bf16)f0[1]; a[2]  = (__bf16)f0[2]; a[3]  = (__bf16)f0[3];
  a[4]  = (__bf16)f1[0]; a[5]  = (__bf16)f1[1]; a[6]  = (__bf16)f1[2]; a[7]  = (__bf16)f1[3];
  a[8]  = (__bf16)f2[0]; a[9]  = (__bf16)f2[1]; a[10] = (__bf16)f2[2]; a[11] = (__bf16)f2[3];
  a[12] = (__bf16)f3[0]; a[13] = (__bf16)f3[1]; a[14] = (__bf16)f3[2]; a[15] = (__bf16)f3[3];
  return a;
}

__device__ __forceinline__ v16bf load_a_f32(const float* __restrict__ row, int kb) {
  return pack_a(*(const v4f*)(row + kb),      *(const v4f*)(row + kb + 4),
                *(const v4f*)(row + kb + 8), *(const v4f*)(row + kb + 12));
}

// A operand = bf16(row_x[k] + row_a[k])  (fused residual-add prologue)
__device__ __forceinline__ v16bf load_a_sum(const float* __restrict__ rx,
                                            const float* __restrict__ ra, int kb) {
  v4f s0 = *(const v4f*)(rx + kb)      + *(const v4f*)(ra + kb);
  v4f s1 = *(const v4f*)(rx + kb + 4)  + *(const v4f*)(ra + kb + 4);
  v4f s2 = *(const v4f*)(rx + kb + 8) + *(const v4f*)(ra + kb + 8);
  v4f s3 = *(const v4f*)(rx + kb + 12) + *(const v4f*)(ra + kb + 12);
  return pack_a(s0, s1, s2, s3);
}

__device__ __forceinline__ v16bf load_a_bf16(const __bf16* __restrict__ row, int kb) {
  v8bf lo = *(const v8bf*)(row + kb);
  v8bf hi = *(const v8bf*)(row + kb + 8);
  return __builtin_shufflevector(lo, hi, 0, 1, 2, 3, 4, 5, 6, 7,
                                 8, 9, 10, 11, 12, 13, 14, 15);
}

// ---------------------------------------------------------------------------
// Fused edge kernel: e = edge_attr @ W_edge + b_edge ; msg = relu(x[src] + e);
// agg[dst] += msg.  One wave = 16 edges; 8 N-tiles x 2 K-steps of WMMA.
// EDGE_BF16: A comes from the bf16 edge cache (half the HBM bytes).
// ---------------------------------------------------------------------------
template <bool EDGE_BF16>
__global__ void __launch_bounds__(256) k_edge_message(
    const float* __restrict__ edge_f32,    // [E, 64] (used if !EDGE_BF16)
    const __bf16* __restrict__ edge_b16,   // [E, 64] (used if EDGE_BF16)
    const int* __restrict__ src, const int* __restrict__ dst,
    const __bf16* __restrict__ x_bf,       // [N, 128]
    const float* __restrict__ W_edge,      // [64, 128]
    const float* __restrict__ b_edge,      // [128]
    float* __restrict__ agg)               // [N, 128]
{
  __shared__ __bf16 ldsW[DIN * DEDGE];     // transposed [n][k], 16 KB
  for (int idx = threadIdx.x; idx < DEDGE * DIN; idx += 256) {
    int k = idx >> 7, n = idx & 127;
    ldsW[n * DEDGE + k] = (__bf16)W_edge[idx];
  }
  __syncthreads();

  int lane = threadIdx.x & 31;
  int wave = threadIdx.x >> 5;
  int tile = blockIdx.x * 8 + wave;
  if (tile >= NEDGES / 16) return;

  int m   = lane & 15;
  int kb  = (lane < 16) ? 0 : 16;   // A-layout K base
  int mo  = (lane < 16) ? 0 : 8;    // C-layout M offset
  int kb2 = (lane < 16) ? 0 : 16;   // B-layout K base

  // Prefetch gather rows early (global_prefetch_b8): 16 distinct x rows/tile.
  {
    int pr = src[tile * 16 + m];
    __builtin_prefetch(x_bf + pr * DIN, 0, 0);
    __builtin_prefetch(x_bf + pr * DIN + 64, 0, 0);
  }

  v16bf a0, a1;
  if (EDGE_BF16) {
    const __bf16* arow = edge_b16 + (tile * 16 + m) * DEDGE;
    a0 = load_a_bf16(arow, kb);
    a1 = load_a_bf16(arow, kb + 32);
  } else {
    const float* arow = edge_f32 + (tile * 16 + m) * DEDGE;
    a0 = load_a_f32(arow, kb);
    a1 = load_a_f32(arow, kb + 32);
  }

  int soff[8], doff[8];
#pragma unroll
  for (int r = 0; r < 8; ++r) {
    int e = tile * 16 + mo + r;
    soff[r] = src[e] * DIN;
    doff[r] = dst[e] * DIN;
  }

#pragma unroll
  for (int j = 0; j < 8; ++j) {
    int n = j * 16 + m;
    const __bf16* wp = ldsW + n * DEDGE + kb2;
    v8f c = {};
    c = wmma_bf16(a0, *(const v16bf*)(wp), c);
    c = wmma_bf16(a1, *(const v16bf*)(wp + 32), c);
    float bias = b_edge[n];
#pragma unroll
    for (int r = 0; r < 8; ++r) {
      float xv  = (float)x_bf[soff[r] + n];         // L2-resident gather
      float msg = c[r] + bias + xv;
      msg = msg > 0.f ? msg : 0.f;
      atomic_add_f32(&agg[doff[r] + n], msg);       // L2-resident scatter
    }
  }
}

// ---------------------------------------------------------------------------
// GEMM1 (fused): h1 = bf16(relu(BN((x_f + agg) @ W1 + b1)))
// ---------------------------------------------------------------------------
__global__ void __launch_bounds__(256) k_gemm_bn_relu(
    const float* __restrict__ x_f, const float* __restrict__ agg,
    const float* __restrict__ W, const float* __restrict__ bias,
    const float* __restrict__ gamma, const float* __restrict__ beta,
    const float* __restrict__ rm, const float* __restrict__ rv,
    __bf16* __restrict__ out)
{
  __shared__ __bf16 ldsW[DIN * DIN];       // transposed [n][k], 32 KB
  for (int idx = threadIdx.x; idx < DIN * DIN; idx += 256) {
    int k = idx >> 7, n = idx & 127;
    ldsW[n * DIN + k] = (__bf16)W[idx];
  }
  __syncthreads();

  int lane = threadIdx.x & 31;
  int wave = threadIdx.x >> 5;
  int tile = blockIdx.x * 8 + wave;
  if (tile >= NNODES / 16) return;

  int m = lane & 15;
  int kb  = (lane < 16) ? 0 : 16;
  int mo  = (lane < 16) ? 0 : 8;
  int kb2 = (lane < 16) ? 0 : 16;

  const float* rx = x_f + (tile * 16 + m) * DIN;
  const float* ra = agg + (tile * 16 + m) * DIN;
  v16bf a0 = load_a_sum(rx, ra, kb);
  v16bf a1 = load_a_sum(rx, ra, kb + 32);
  v16bf a2 = load_a_sum(rx, ra, kb + 64);
  v16bf a3 = load_a_sum(rx, ra, kb + 96);

#pragma unroll
  for (int j = 0; j < 8; ++j) {
    int n = j * 16 + m;
    const __bf16* wp = ldsW + n * DIN + kb2;
    v8f c = {};
    c = wmma_bf16(a0, *(const v16bf*)(wp), c);
    c = wmma_bf16(a1, *(const v16bf*)(wp + 32), c);
    c = wmma_bf16(a2, *(const v16bf*)(wp + 64), c);
    c = wmma_bf16(a3, *(const v16bf*)(wp + 96), c);
    float bb = bias[n];
    float sc = gamma[n] * rsqrtf(rv[n] + BN_EPS);
    float sh = beta[n] - rm[n] * sc;
#pragma unroll
    for (int r = 0; r < 8; ++r) {
      float v = (c[r] + bb) * sc + sh;
      v = v > 0.f ? v : 0.f;
      out[(tile * 16 + mo + r) * DIN + n] = (__bf16)v;
    }
  }
}

// ---------------------------------------------------------------------------
// GEMM2: y = A @ W2 + b2.
// mode 0 (inner layers): y = relu(y); write f32 x + bf16 mirror.
// mode 1 (last layer):   scatter y into pool sums[batch[row]] via f32 atomics.
// ---------------------------------------------------------------------------
__global__ void __launch_bounds__(256) k_gemm_out(
    const __bf16* __restrict__ A, const float* __restrict__ W,
    const float* __restrict__ bias,
    float* __restrict__ xf, __bf16* __restrict__ xb,
    const int* __restrict__ batch, float* __restrict__ sums, int mode)
{
  __shared__ __bf16 ldsW[DIN * DIN];
  for (int idx = threadIdx.x; idx < DIN * DIN; idx += 256) {
    int k = idx >> 7, n = idx & 127;
    ldsW[n * DIN + k] = (__bf16)W[idx];
  }
  __syncthreads();

  int lane = threadIdx.x & 31;
  int wave = threadIdx.x >> 5;
  int tile = blockIdx.x * 8 + wave;
  if (tile >= NNODES / 16) return;

  int m = lane & 15;
  int kb  = (lane < 16) ? 0 : 16;
  int mo  = (lane < 16) ? 0 : 8;
  int kb2 = (lane < 16) ? 0 : 16;

  const __bf16* arow = A + (tile * 16 + m) * DIN;
  v16bf a0 = load_a_bf16(arow, kb);
  v16bf a1 = load_a_bf16(arow, kb + 32);
  v16bf a2 = load_a_bf16(arow, kb + 64);
  v16bf a3 = load_a_bf16(arow, kb + 96);

  int goff[8];
  if (mode == 1) {
#pragma unroll
    for (int r = 0; r < 8; ++r) goff[r] = batch[tile * 16 + mo + r] * DIN;
  }

#pragma unroll
  for (int j = 0; j < 8; ++j) {
    int n = j * 16 + m;
    const __bf16* wp = ldsW + n * DIN + kb2;
    v8f c = {};
    c = wmma_bf16(a0, *(const v16bf*)(wp), c);
    c = wmma_bf16(a1, *(const v16bf*)(wp + 32), c);
    c = wmma_bf16(a2, *(const v16bf*)(wp + 64), c);
    c = wmma_bf16(a3, *(const v16bf*)(wp + 96), c);
    float bb = bias[n];
    if (mode == 1) {
#pragma unroll
      for (int r = 0; r < 8; ++r)
        atomic_add_f32(&sums[goff[r] + n], c[r] + bb);   // fused mean-pool
    } else {
#pragma unroll
      for (int r = 0; r < 8; ++r) {
        float v = c[r] + bb;
        v = v > 0.f ? v : 0.f;
        int o = (tile * 16 + mo + r) * DIN + n;
        xf[o] = v;
        xb[o] = (__bf16)v;
      }
    }
  }
}

// ---------------------------------------------------------------------------
// Elementwise / pooling helpers
// ---------------------------------------------------------------------------
__global__ void __launch_bounds__(256) k_init_x(const float* __restrict__ x,
                                                float* __restrict__ xf,
                                                __bf16* __restrict__ xb, int n) {
  int i = blockIdx.x * 256 + threadIdx.x;
  if (i < n) { float v = x[i]; xf[i] = v; xb[i] = (__bf16)v; }
}

__global__ void __launch_bounds__(256) k_cvt_edge(const float* __restrict__ e,
                                                  __bf16* __restrict__ eb, int n4) {
  int i = blockIdx.x * 256 + threadIdx.x;   // one float4 per thread
  if (i < n4) {
    v4f f = ((const v4f*)e)[i];
    v8bf* o = (v8bf*)eb;                    // write 4 bf16 as half of v8bf slot
    __bf16* p = (__bf16*)(o) + i * 4;
    p[0] = (__bf16)f[0]; p[1] = (__bf16)f[1];
    p[2] = (__bf16)f[2]; p[3] = (__bf16)f[3];
  }
}

__global__ void __launch_bounds__(256) k_counts(const int* __restrict__ batch,
                                                float* __restrict__ counts, int n) {
  int i = blockIdx.x * 256 + threadIdx.x;
  if (i < n) atomic_add_f32(&counts[batch[i]], 1.0f);
}

__global__ void __launch_bounds__(128) k_div(const float* __restrict__ sums,
                                             const float* __restrict__ counts,
                                             float* __restrict__ out) {
  int g = blockIdx.x, f = threadIdx.x;
  float c = counts[g];
  c = c > 1.f ? c : 1.f;
  out[g * DIN + f] = sums[g * DIN + f] / c;
}

// ---------------------------------------------------------------------------
extern "C" void kernel_launch(void* const* d_in, const int* in_sizes, int n_in,
                              void* d_out, int out_size, void* d_ws, size_t ws_size,
                              hipStream_t stream) {
  const float* x         = (const float*)d_in[0];
  const int*   edge_idx  = (const int*)d_in[1];
  const float* edge_attr = (const float*)d_in[2];
  const int*   batch     = (const int*)d_in[3];
  const float* W_edge    = (const float*)d_in[4];
  const float* b_edge    = (const float*)d_in[5];
  const float* W1        = (const float*)d_in[6];
  const float* b1        = (const float*)d_in[7];
  const float* gamma     = (const float*)d_in[8];
  const float* beta      = (const float*)d_in[9];
  const float* rm        = (const float*)d_in[10];
  const float* rv        = (const float*)d_in[11];
  const float* W2        = (const float*)d_in[12];
  const float* b2        = (const float*)d_in[13];

  const int* src = edge_idx;
  const int* dst = edge_idx + NEDGES;

  char* ws = (char*)d_ws;
  size_t off = 0;
  auto alloc = [&](size_t bytes) {
    void* p = ws + off;
    off = (off + bytes + 255) & ~(size_t)255;
    return p;
  };
  float*  x_f    = (float*)alloc((size_t)NF * 4);
  float*  agg    = (float*)alloc((size_t)NF * 4);
  __bf16* x_b    = (__bf16*)alloc((size_t)NF * 2);
  __bf16* h1_b   = (__bf16*)alloc((size_t)NF * 2);
  float*  sums   = (float*)alloc((size_t)NGRAPHS * DIN * 4);
  float*  counts = (float*)alloc((size_t)NGRAPHS * 4);

  // Optional bf16 edge-attr cache (halves edge HBM traffic): gate on ws_size.
  const size_t edge_bytes = (size_t)NEDGES * DEDGE * 2;
  bool use_edge_cache = (ws_size >= off + edge_bytes + 4096);
  __bf16* edge_b = use_edge_cache ? (__bf16*)alloc(edge_bytes) : nullptr;

  const int ew_blocks = (NEDGES / 16 + 7) / 8;   // 4688
  const int nd_blocks = (NNODES / 16 + 7) / 8;   // 391
  const int el_blocks = (NF + 255) / 256;

  k_init_x<<<el_blocks, 256, 0, stream>>>(x, x_f, x_b, NF);
  hipMemsetAsync(sums, 0, (size_t)NGRAPHS * DIN * 4, stream);
  hipMemsetAsync(counts, 0, (size_t)NGRAPHS * 4, stream);
  k_counts<<<(NNODES + 255) / 256, 256, 0, stream>>>(batch, counts, NNODES);
  if (use_edge_cache) {
    int n4 = NEDGES * DEDGE / 4;
    k_cvt_edge<<<(n4 + 255) / 256, 256, 0, stream>>>(edge_attr, edge_b, n4);
  }

  for (int i = 0; i < NLAYERS; ++i) {
    hipMemsetAsync(agg, 0, (size_t)NF * 4, stream);
    if (use_edge_cache) {
      k_edge_message<true><<<ew_blocks, 256, 0, stream>>>(
          edge_attr, edge_b, src, dst, x_b,
          W_edge + (size_t)i * DEDGE * DIN, b_edge + (size_t)i * DIN, agg);
    } else {
      k_edge_message<false><<<ew_blocks, 256, 0, stream>>>(
          edge_attr, edge_b, src, dst, x_b,
          W_edge + (size_t)i * DEDGE * DIN, b_edge + (size_t)i * DIN, agg);
    }
    k_gemm_bn_relu<<<nd_blocks, 256, 0, stream>>>(
        x_f, agg, W1 + (size_t)i * DIN * DIN, b1 + (size_t)i * DIN,
        gamma + (size_t)i * DIN, beta + (size_t)i * DIN,
        rm + (size_t)i * DIN, rv + (size_t)i * DIN, h1_b);
    k_gemm_out<<<nd_blocks, 256, 0, stream>>>(
        h1_b, W2 + (size_t)i * DIN * DIN, b2 + (size_t)i * DIN,
        x_f, x_b, batch, sums, (i == NLAYERS - 1) ? 1 : 0);
  }

  k_div<<<NGRAPHS, 128, 0, stream>>>(sums, counts, (float*)d_out);
}
